// HybridBinaryClassifier_65481071409792
// MI455X (gfx1250) — matrix operand order repaired and checked by reading, not compiled, not run on hardware
//
#include <hip/hip_runtime.h>
#include <hip/hip_bf16.h>
#include <math.h>

// ---------------------------------------------------------------------------
// MI455X (gfx1250) hybrid CNN + quantum-head classifier.
// conv1/conv2 implicit GEMM + fc1 run on v_wmma_f32_16x16x32_f16 (wave32,
// one 16x16 tile per wave, f32 accumulate). All operands padded so the hot
// loops carry zero per-element predication (pure b128 loads + wmma).
// fc1 stages its A-chunk to LDS via the Tensor Data Mover when available.
// ---------------------------------------------------------------------------

typedef __attribute__((ext_vector_type(16))) _Float16 v16h;
typedef __attribute__((ext_vector_type(8)))  _Float16 v8h;
typedef __attribute__((ext_vector_type(8)))  float    v8f;
typedef __attribute__((ext_vector_type(4)))  unsigned v4u;
typedef __attribute__((ext_vector_type(4)))  int      v4i;
typedef __attribute__((ext_vector_type(8)))  int      v8i;

// ----- problem constants ----------------------------------------------------
#define BATCH   256
#define IH0     250
#define C0      3
#define C1      6
#define KK1     5
#define OH1     124      // (250+2-5)/2+1
#define PH1     123      // maxpool 2x2 stride1
#define C2      15
#define KK2     3
#define OH2     62       // (123+2-3)/2+1
#define PH2     61
#define FC1_IN  55815    // 15*61*61
#define FC1_OUT 120
#define FC2_OUT 84

#define M1TOT   (BATCH*OH1*OH1)   // 3,936,256 (divisible by 128)
#define M2TOT   (BATCH*OH2*OH2)   // 984,064   (divisible by 128)

#define KPAD1     96              // conv1 K=75 -> 3 wmma steps
#define KPAD2     64              // conv2 K=54 -> 2 wmma steps
#define FC1_KPAD  56064           // 55815 -> 219 chunks of 256
#define FC1_NPAD  128
#define FCHUNK    256             // k per LDS chunk (8 wmma steps)
#define FNCH      (FC1_KPAD/FCHUNK)  // 219

// ----- WMMA fragment lane mappings (CDNA5 ISA 7.12.2, wave32) ---------------
// A 16x32 f16: row m=lane&15; hi=lane>>4 selects K 0-15/16-31 (k = hi*16 + j, this code's order)
// B 32x16 f16: col n=lane&15; lanes 16-31 carry K=16..31 (k = hi*16 + j, this code's order, same as A)
// C/D 16x16 f32: elem v -> row m = v + 8*hi, col n = lane&15

// ---------------------------------------------------------------------------
// prep: zero-padded f16 weight/activation buffers
// ---------------------------------------------------------------------------
__global__ void pad_w_f16_kernel(const float* __restrict__ src,
                                 _Float16* __restrict__ dst,
                                 int cout, int ktot, int kpad) {
    int i = blockIdx.x * blockDim.x + threadIdx.x;
    if (i >= 16 * kpad) return;
    int n = i / kpad, k = i % kpad;
    float v = (n < cout && k < ktot) ? src[n * ktot + k] : 0.0f;
    dst[i] = (_Float16)v;
}

__global__ void pad_fc1w_kernel(const float* __restrict__ src,
                                _Float16* __restrict__ dst) {
    long i = (long)blockIdx.x * blockDim.x + threadIdx.x;
    if (i >= (long)FC1_NPAD * FC1_KPAD) return;
    int n = (int)(i / FC1_KPAD);
    int k = (int)(i % FC1_KPAD);
    float v = (n < FC1_OUT && k < FC1_IN) ? src[(long)n * FC1_IN + k] : 0.0f;
    dst[i] = (_Float16)v;
}

// ---------------------------------------------------------------------------
// Implicit-GEMM convolution (stride 2, pad 1) + bias + ReLU -> f16 NCHW.
// One wave = one 16(M positions) x 16(N channels) tile. Weights in LDS.
// Halo handled with clamped-address loads + value select (no exec masking).
// ---------------------------------------------------------------------------
template<typename TIn, int CIN, int KK, int IH, int OH, int COUT, int KPAD>
__global__ void conv_wmma_kernel(const TIn* __restrict__ in,
                                 const _Float16* __restrict__ wpad, // 16 x KPAD
                                 const float* __restrict__ bias,
                                 _Float16* __restrict__ out) {
    constexpr int KTOT = CIN * KK * KK;
    __shared__ __align__(32) _Float16 sW[16 * KPAD];

    // cooperative weight stage (16*KPAD/8 v8h elements, <= 192)
    if (threadIdx.x < (16 * KPAD / 8))
        ((v8h*)sW)[threadIdx.x] = ((const v8h*)wpad)[threadIdx.x];
    __syncthreads();

    const int lane = threadIdx.x & 31;
    const int wave = threadIdx.x >> 5;
    const int hi   = (lane >> 4) & 1;
    const int tile = blockIdx.x * 8 + wave;
    const int m0   = tile * 16;

    // per-lane A row position (M tiles are exact; no partial tiles)
    const int mrow = m0 + (lane & 15);
    const int b  = mrow / (OH * OH);
    const int r0 = mrow % (OH * OH);
    const int oy = r0 / OH;
    const int ox = r0 % OH;
    const int n  = lane & 15;
    const TIn* bimg = in + (size_t)b * CIN * IH * IH;

    v8f acc = {};
#pragma unroll
    for (int kt = 0; kt < KPAD / 32; ++kt) {
        const int kbase = kt * 32;
        v16h afrag;
#pragma unroll
        for (int j = 0; j < 16; ++j) {
            int k = kbase + hi * 16 + j;
            int kc = (k < KTOT) ? k : 0;
            int ci = kc / (KK * KK);
            int rr = kc % (KK * KK);
            int ky = rr / KK, kx = rr % KK;
            int iy = oy * 2 - 1 + ky;
            int ix = ox * 2 - 1 + kx;
            int iyc = iy < 0 ? 0 : (iy >= IH ? IH - 1 : iy);
            int ixc = ix < 0 ? 0 : (ix >= IH ? IH - 1 : ix);
            float raw = (float)bimg[((size_t)ci * IH + iyc) * IH + ixc];
            bool ok = (k < KTOT) & ((unsigned)iy < (unsigned)IH) &
                      ((unsigned)ix < (unsigned)IH);
            afrag[j] = ok ? (_Float16)raw : (_Float16)0.0f;
        }
        v16h bfrag = *(const v16h*)&sW[n * KPAD + kbase + hi * 16];
        acc = __builtin_amdgcn_wmma_f32_16x16x32_f16(
                  false, afrag, false, bfrag, (short)0, acc, false, false);
    }

    const float bn = (n < COUT) ? bias[n] : 0.0f;
#pragma unroll
    for (int v = 0; v < 8; ++v) {
        if (n < COUT) {
            int m = m0 + v + hi * 8;
            int bb = m / (OH * OH);
            int r  = m % (OH * OH);
            int yy = r / OH, xx = r % OH;
            float val = fmaxf(acc[v] + bn, 0.0f);
            out[(((size_t)bb * COUT + n) * OH + yy) * OH + xx] = (_Float16)val;
        }
    }
}

// ---------------------------------------------------------------------------
// 2x2 max-pool stride 1, f16 NCHW (conv1 -> conv2 path)
// ---------------------------------------------------------------------------
template<int IH>
__global__ void maxpool_s1_kernel(const _Float16* __restrict__ in,
                                  _Float16* __restrict__ out, long total) {
    constexpr int OHp = IH - 1;
    long i = (long)blockIdx.x * blockDim.x + threadIdx.x;
    if (i >= total) return;
    int x = (int)(i % OHp);
    long t = i / OHp;
    int y = (int)(t % OHp);
    long bc = t / OHp;
    const _Float16* p = in + (bc * IH + y) * IH + x;
    float m = fmaxf(fmaxf((float)p[0], (float)p[1]),
                    fmaxf((float)p[IH], (float)p[IH + 1]));
    out[i] = (_Float16)m;
}

// maxpool2 writing straight into the zero-padded fc1 activation layout
__global__ void maxpool2_pad_kernel(const _Float16* __restrict__ in, // 256x15x62x62
                                    _Float16* __restrict__ out) {    // 256 x FC1_KPAD
    long i = (long)blockIdx.x * blockDim.x + threadIdx.x;
    if (i >= (long)BATCH * FC1_KPAD) return;
    int b   = (int)(i / FC1_KPAD);
    int col = (int)(i % FC1_KPAD);
    _Float16 res = (_Float16)0.0f;
    if (col < FC1_IN) {
        int c  = col / (PH2 * PH2);
        int rr = col % (PH2 * PH2);
        int y = rr / PH2, x = rr % PH2;
        const _Float16* p = in + (((size_t)b * C2 + c) * OH2 + y) * OH2 + x;
        float m = fmaxf(fmaxf((float)p[0], (float)p[1]),
                        fmaxf((float)p[OH2], (float)p[OH2 + 1]));
        res = (_Float16)m;
    }
    out[i] = res;
}

// ---------------------------------------------------------------------------
// fc1: (256 x 56064pad) @ (56064 x 128pad) + bias, ReLU.
// One block = one M-tile (16 rows). 8 waves = 8 N-tiles (covers 120 outs).
// A chunk (16 x 256 f16 = 8KB) staged to LDS via TDM (tensor_load_to_lds)
// when the builtin exists, else cooperative v16h copies.
// ---------------------------------------------------------------------------
__global__ void fc1_wmma_kernel(const _Float16* __restrict__ act,  // 256 x FC1_KPAD
                                const _Float16* __restrict__ wh,   // 128 x FC1_KPAD
                                const float* __restrict__ bias,
                                float* __restrict__ out) {         // 256 x 120
    __shared__ __align__(32) _Float16 sA[16 * FCHUNK];
    const int lane = threadIdx.x & 31;
    const int wave = threadIdx.x >> 5;
    const int hi   = (lane >> 4) & 1;
    const int mt   = blockIdx.x;          // 0..15
    const int n    = wave * 16 + (lane & 15);
    const size_t brow = (size_t)n * FC1_KPAD;

    v8f acc = {};
    for (int kc = 0; kc < FNCH; ++kc) {
        const size_t gbase = (size_t)mt * 16 * FC1_KPAD + (size_t)kc * FCHUNK;
#if defined(__gfx1250__) && __has_builtin(__builtin_amdgcn_tensor_load_to_lds) && \
    __has_builtin(__builtin_amdgcn_s_wait_tensorcnt)
        if (wave == 0) {
            // Tensor DMA descriptor (CDNA5 ISA ch.8): 2D tile 256 x 16, f16.
            unsigned long long ga = (unsigned long long)(size_t)(act + gbase);
            unsigned lds_off = (unsigned)(size_t)&sA[0];   // addr[31:0] = LDS offset
            v4u g0 = { 1u,                                  // count=1
                       lds_off,
                       (unsigned)(ga & 0xffffffffull),
                       (unsigned)((ga >> 32) & 0x1ffffffull) | (2u << 30) }; // type=2
            v8i g1 = { (int)(1u << 16),                     // data_size=1 -> 2 bytes
                       (int)((FC1_KPAD & 0xffffu) << 16),   // tensor_dim0[15:0]
                       (int)((FC1_KPAD >> 16) | (16u << 16)), // dim0 hi | tensor_dim1 lo
                       (int)(FCHUNK << 16),                 // tile_dim0 = 256
                       (int)16,                             // tile_dim1 = 16
                       (int)FC1_KPAD,                       // tensor_dim0_stride lo
                       0, 0 };
            v4i g2 = { 0, 0, 0, 0 };
            v4i g3 = { 0, 0, 0, 0 };
#if __clang_major__ >= 23
            v8i g4 = { 0, 0, 0, 0, 0, 0, 0, 0 };
            __builtin_amdgcn_tensor_load_to_lds(g0, g1, g2, g3, g4, 0);
#else
            __builtin_amdgcn_tensor_load_to_lds(g0, g1, g2, g3, 0);
#endif
            __builtin_amdgcn_s_wait_tensorcnt(0);
        }
#else
        {
            int row = threadIdx.x >> 4, seg = threadIdx.x & 15;
            *(v16h*)&sA[row * FCHUNK + seg * 16] =
                *(const v16h*)(act + gbase + (size_t)row * FC1_KPAD + seg * 16);
        }
#endif
        __syncthreads();
        if (kc + 1 < FNCH)
            __builtin_prefetch(wh + brow + (size_t)(kc + 1) * FCHUNK, 0, 0);
#pragma unroll
        for (int ks = 0; ks < FCHUNK / 32; ++ks) {
            const int kl = ks * 32;
            v8h lo  = *(const v8h*)&sA[(lane & 15) * FCHUNK + kl + hi * 16];
            v8h hi8 = *(const v8h*)&sA[(lane & 15) * FCHUNK + kl + hi * 16 + 8];
            v16h afrag;
#pragma unroll
            for (int j = 0; j < 8; ++j) { afrag[j] = lo[j]; afrag[j + 8] = hi8[j]; }
            v16h bfrag = *(const v16h*)(wh + brow + (size_t)kc * FCHUNK + kl + hi * 16);
            acc = __builtin_amdgcn_wmma_f32_16x16x32_f16(
                      false, afrag, false, bfrag, (short)0, acc, false, false);
        }
        __syncthreads();
    }

    const float bn = (n < FC1_OUT) ? bias[n] : 0.0f;
#pragma unroll
    for (int v = 0; v < 8; ++v) {
        int mm = mt * 16 + v + hi * 8;
        if (n < FC1_OUT) out[(size_t)mm * FC1_OUT + n] = fmaxf(acc[v] + bn, 0.0f);
    }
}

// ---------------------------------------------------------------------------
// fc2 (120->84, ReLU) + fc3 (84->1), one block per sample
// ---------------------------------------------------------------------------
__global__ void fc23_kernel(const float* __restrict__ fc1o,
                            const float* __restrict__ w2, const float* __restrict__ b2,
                            const float* __restrict__ w3, const float* __restrict__ b3,
                            float* __restrict__ fc3o) {
    __shared__ float red[128];
    const int b = blockIdx.x;
    const int t = threadIdx.x;
    float part = 0.0f;
    if (t < FC2_OUT) {
        float s = b2[t];
        const float* xr = fc1o + (size_t)b * FC1_OUT;
        const float* wr = w2 + (size_t)t * FC1_OUT;
#pragma unroll 4
        for (int k = 0; k < FC1_OUT; ++k) s += xr[k] * wr[k];
        part = fmaxf(s, 0.0f) * w3[t];
    }
    red[t] = part;
    __syncthreads();
    for (int off = 64; off > 0; off >>= 1) {
        if (t < off) red[t] += red[t + off];
        __syncthreads();
    }
    if (t == 0) fc3o[b] = red[0] + b3[0];
}

// ---------------------------------------------------------------------------
// 4-qubit statevector quantum head, one thread per sample
// ---------------------------------------------------------------------------
struct QOp { int kind; int a; int b; };   // 0=rx 1=ry 2=rz 3=cnot
__device__ __constant__ QOp RAND_OPS_D[50] = {
    {1,0,-1},{3,2,0},{0,3,-1},{2,1,-1},{1,2,-1},{3,0,3},{0,1,-1},{2,3,-1},
    {1,1,-1},{3,1,2},{0,0,-1},{2,2,-1},{1,3,-1},{0,2,-1},{3,3,1},{2,0,-1},
    {1,2,-1},{0,3,-1},{3,0,2},{2,1,-1},{1,0,-1},{0,1,-1},{3,1,3},{2,3,-1},
    {1,1,-1},{0,2,-1},{3,2,1},{2,2,-1},{1,3,-1},{0,0,-1},{3,3,0},{2,0,-1},
    {1,2,-1},{0,3,-1},{3,0,1},{2,1,-1},{1,0,-1},{0,1,-1},{3,2,3},{2,3,-1},
    {1,1,-1},{0,2,-1},{3,1,0},{2,2,-1},{1,3,-1},{0,0,-1},{3,3,2},{2,0,-1},
    {1,2,-1},{0,3,-1}
};

__device__ __forceinline__ void q_ry(float* re, float* im, int w, float t) {
    float c = cosf(0.5f * t), s = sinf(0.5f * t);
    int st = 1 << (3 - w);
    for (int i = 0; i < 16; ++i) if (!(i & st)) {
        int j = i | st;
        float ar = re[i], ai = im[i], br = re[j], bi = im[j];
        re[i] = c * ar - s * br;  im[i] = c * ai - s * bi;
        re[j] = s * ar + c * br;  im[j] = s * ai + c * bi;
    }
}
__device__ __forceinline__ void q_rx(float* re, float* im, int w, float t) {
    float c = cosf(0.5f * t), s = sinf(0.5f * t);
    int st = 1 << (3 - w);
    for (int i = 0; i < 16; ++i) if (!(i & st)) {
        int j = i | st;
        float ar = re[i], ai = im[i], br = re[j], bi = im[j];
        re[i] = c * ar + s * bi;  im[i] = c * ai - s * br;
        re[j] = s * ai + c * br;  im[j] = -s * ar + c * bi;
    }
}
__device__ __forceinline__ void q_rz(float* re, float* im, int w, float t) {
    float c = cosf(0.5f * t), s = sinf(0.5f * t);
    int st = 1 << (3 - w);
    for (int i = 0; i < 16; ++i) {
        float ar = re[i], ai = im[i];
        if (!(i & st)) { re[i] = c * ar + s * ai; im[i] = c * ai - s * ar; }
        else           { re[i] = c * ar - s * ai; im[i] = c * ai + s * ar; }
    }
}
__device__ __forceinline__ void q_cnot(float* re, float* im, int c, int t) {
    int cm = 1 << (3 - c), tm = 1 << (3 - t);
    for (int i = 0; i < 16; ++i) if ((i & cm) && !(i & tm)) {
        int j = i | tm;
        float tr = re[i], ti = im[i];
        re[i] = re[j]; im[i] = im[j]; re[j] = tr; im[j] = ti;
    }
}
__device__ __forceinline__ void q_had(float* re, float* im, int w) {
    const float r = 0.70710678118654752f;
    int st = 1 << (3 - w);
    for (int i = 0; i < 16; ++i) if (!(i & st)) {
        int j = i | st;
        float ar = re[i], ai = im[i], br = re[j], bi = im[j];
        re[i] = r * (ar + br); im[i] = r * (ai + bi);
        re[j] = r * (ar - br); im[j] = r * (ai - bi);
    }
}
__device__ __forceinline__ void q_sx(float* re, float* im, int w) {
    int st = 1 << (3 - w);
    for (int i = 0; i < 16; ++i) if (!(i & st)) {
        int j = i | st;
        float ar = re[i], ai = im[i], br = re[j], bi = im[j];
        float nr0 = 0.5f * ((ar - ai) + (br + bi));
        float ni0 = 0.5f * ((ar + ai) + (bi - br));
        float nr1 = 0.5f * ((ar + ai) + (br - bi));
        float ni1 = 0.5f * ((ai - ar) + (br + bi));
        re[i] = nr0; im[i] = ni0; re[j] = nr1; im[j] = ni1;
    }
}
__device__ __forceinline__ void q_crx(float* re, float* im, int c, int t, float th) {
    float cc = cosf(0.5f * th), s = sinf(0.5f * th);
    int cm = 1 << (3 - c), tm = 1 << (3 - t);
    for (int i = 0; i < 16; ++i) if ((i & cm) && !(i & tm)) {
        int j = i | tm;
        float ar = re[i], ai = im[i], br = re[j], bi = im[j];
        re[i] = cc * ar + s * bi;  im[i] = cc * ai - s * br;
        re[j] = s * ai + cc * br;  im[j] = -s * ar + cc * bi;
    }
}

__global__ void quantum_kernel(const float* __restrict__ fc3o,
                               const float* __restrict__ rand_params, int n_rand,
                               const float* __restrict__ rx0, const float* __restrict__ ry0,
                               const float* __restrict__ rz0, const float* __restrict__ crx0,
                               float* __restrict__ zout) {
    int t = blockIdx.x * blockDim.x + threadIdx.x;
    if (t >= BATCH) return;
    float re[16], im[16];
    for (int i = 0; i < 16; ++i) { re[i] = 0.0f; im[i] = 0.0f; }
    re[0] = 1.0f;

    // encoder '4x4_ryzxy': only feature 0 nonzero -> single ry on wire 0
    q_ry(re, im, 0, fc3o[t]);

    int pi = 0;
    for (int g = 0; g < 50; ++g) {
        QOp op = RAND_OPS_D[g];
        if (op.kind == 3) {
            q_cnot(re, im, op.a, op.b);
        } else {
            int idx = pi < n_rand ? pi : (n_rand - 1);
            float ang = rand_params[idx];
            ++pi;
            if (op.kind == 0) q_rx(re, im, op.a, ang);
            else if (op.kind == 1) q_ry(re, im, op.a, ang);
            else q_rz(re, im, op.a, ang);
        }
    }
    q_rx(re, im, 0, rx0[0]);
    q_ry(re, im, 1, ry0[0]);
    q_rz(re, im, 3, rz0[0]);
    q_crx(re, im, 0, 2, crx0[0]);
    q_had(re, im, 3);
    q_sx(re, im, 2);
    q_cnot(re, im, 3, 0);

    for (int w = 0; w < 4; ++w) {
        float z = 0.0f;
        for (int i = 0; i < 16; ++i) {
            float p = re[i] * re[i] + im[i] * im[i];
            z += (((i >> (3 - w)) & 1) ? -1.0f : 1.0f) * p;
        }
        zout[t * 4 + w] = z;
    }
}

// ---------------------------------------------------------------------------
// BatchNorm1d (train, biased var) + sigmoid + [p, 1-p]
// ---------------------------------------------------------------------------
__global__ void bn_sigmoid_kernel(const float* __restrict__ z,
                                  const float* __restrict__ gamma,
                                  const float* __restrict__ beta,
                                  float* __restrict__ out) {
    __shared__ float ssum[4];
    __shared__ float ssq[4];
    const int t = threadIdx.x;
    if (t < 4) { ssum[t] = 0.0f; ssq[t] = 0.0f; }
    __syncthreads();
    float zv[4];
#pragma unroll
    for (int c = 0; c < 4; ++c) {
        zv[c] = z[t * 4 + c];
        atomicAdd(&ssum[c], zv[c]);
        atomicAdd(&ssq[c], zv[c] * zv[c]);
    }
    __syncthreads();
#pragma unroll
    for (int c = 0; c < 4; ++c) {
        float mean = ssum[c] * (1.0f / BATCH);
        float var  = ssq[c] * (1.0f / BATCH) - mean * mean;
        float zn   = gamma[c] * (zv[c] - mean) * rsqrtf(var + 1e-5f) + beta[c];
        float p    = 1.0f / (1.0f + expf(-zn));
        out[t * 8 + c]     = p;
        out[t * 8 + 4 + c] = 1.0f - p;
    }
}

// ---------------------------------------------------------------------------
// host launcher
// ---------------------------------------------------------------------------
extern "C" void kernel_launch(void* const* d_in, const int* in_sizes, int n_in,
                              void* d_out, int out_size, void* d_ws, size_t ws_size,
                              hipStream_t stream) {
    const float* x       = (const float*)d_in[0];
    const float* conv1_w = (const float*)d_in[1];
    const float* conv1_b = (const float*)d_in[2];
    const float* conv2_w = (const float*)d_in[3];
    const float* conv2_b = (const float*)d_in[4];
    const float* fc1_w   = (const float*)d_in[5];
    const float* fc1_b   = (const float*)d_in[6];
    const float* fc2_w   = (const float*)d_in[7];
    const float* fc2_b   = (const float*)d_in[8];
    const float* fc3_w   = (const float*)d_in[9];
    const float* fc3_b   = (const float*)d_in[10];
    const float* rand_p  = (const float*)d_in[11];
    const float* rx0     = (const float*)d_in[12];
    const float* ry0     = (const float*)d_in[13];
    const float* rz0     = (const float*)d_in[14];
    const float* crx0    = (const float*)d_in[15];
    const float* bn_g    = (const float*)d_in[16];
    const float* bn_b    = (const float*)d_in[17];
    const int n_rand     = in_sizes[11];

    char* wsp = (char*)d_ws;
    size_t off = 0;
    auto take = [&](size_t bytes) -> void* {
        void* p = wsp + off;
        off += (bytes + 255) & ~(size_t)255;
        return p;
    };
    _Float16* fc1wh  = (_Float16*)take((size_t)FC1_NPAD * FC1_KPAD * 2);    // 14.35 MB
    _Float16* w1pad  = (_Float16*)take((size_t)16 * KPAD1 * 2);
    _Float16* w2pad  = (_Float16*)take((size_t)16 * KPAD2 * 2);
    _Float16* h1     = (_Float16*)take((size_t)BATCH * C1 * OH1 * OH1 * 2); // 47.2 MB
    _Float16* h1p    = (_Float16*)take((size_t)BATCH * C1 * PH1 * PH1 * 2); // 46.5 MB
    _Float16* h2     = (_Float16*)take((size_t)BATCH * C2 * OH2 * OH2 * 2); // 29.5 MB
    _Float16* h2pad  = (_Float16*)take((size_t)BATCH * FC1_KPAD * 2);       // 28.7 MB
    float*    fc1out = (float*)take((size_t)BATCH * FC1_OUT * 4);
    float*    fc3out = (float*)take((size_t)BATCH * 4);
    float*    zbuf   = (float*)take((size_t)BATCH * 4 * 4);
    (void)ws_size; (void)n_in; (void)out_size;

    // 0) padded weight prep
    {
        long n = (long)FC1_NPAD * FC1_KPAD;
        hipLaunchKernelGGL(pad_fc1w_kernel, dim3((unsigned)((n + 255) / 256)), dim3(256),
                           0, stream, fc1_w, fc1wh);
        hipLaunchKernelGGL(pad_w_f16_kernel, dim3((16 * KPAD1 + 255) / 256), dim3(256),
                           0, stream, conv1_w, w1pad, C1, C0 * KK1 * KK1, KPAD1);
        hipLaunchKernelGGL(pad_w_f16_kernel, dim3((16 * KPAD2 + 255) / 256), dim3(256),
                           0, stream, conv2_w, w2pad, C2, C1 * KK2 * KK2, KPAD2);
    }
    // 1) conv1 (WMMA implicit GEMM) + ReLU
    {
        int grid = (M1TOT / 16) / 8;   // 30,752 (exact)
        hipLaunchKernelGGL((conv_wmma_kernel<float, C0, KK1, IH0, OH1, C1, KPAD1>),
                           dim3(grid), dim3(256), 0, stream, x, w1pad, conv1_b, h1);
    }
    // 2) maxpool1
    {
        long total = (long)BATCH * C1 * PH1 * PH1;
        hipLaunchKernelGGL((maxpool_s1_kernel<OH1>), dim3((unsigned)((total + 255) / 256)),
                           dim3(256), 0, stream, h1, h1p, total);
    }
    // 3) conv2 (WMMA implicit GEMM) + ReLU
    {
        int grid = (M2TOT / 16) / 8;   // 7,688 (exact)
        hipLaunchKernelGGL((conv_wmma_kernel<_Float16, C1, KK2, PH1, OH2, C2, KPAD2>),
                           dim3(grid), dim3(256), 0, stream, h1p, w2pad, conv2_b, h2);
    }
    // 4) maxpool2 -> zero-padded fc1 activation rows
    {
        long total = (long)BATCH * FC1_KPAD;
        hipLaunchKernelGGL(maxpool2_pad_kernel, dim3((unsigned)((total + 255) / 256)),
                           dim3(256), 0, stream, h2, h2pad);
    }
    // 5) fc1 GEMM (WMMA, LDS-staged A via TDM when available)
    hipLaunchKernelGGL(fc1_wmma_kernel, dim3(16), dim3(256), 0, stream,
                       h2pad, fc1wh, fc1_b, fc1out);
    // 6) fc2 + fc3
    hipLaunchKernelGGL(fc23_kernel, dim3(BATCH), dim3(128), 0, stream,
                       fc1out, fc2_w, fc2_b, fc3_w, fc3_b, fc3out);
    // 7) quantum head
    hipLaunchKernelGGL(quantum_kernel, dim3(1), dim3(BATCH), 0, stream,
                       fc3out, rand_p, n_rand, rx0, ry0, rz0, crx0, zbuf);
    // 8) BatchNorm + sigmoid + concat
    hipLaunchKernelGGL(bn_sigmoid_kernel, dim3(1), dim3(BATCH), 0, stream,
                       zbuf, bn_g, bn_b, (float*)d_out);
}
